// MixquantLinear_84585085927789
// MI455X (gfx1250) — matrix operand order repaired and checked by reading, not compiled, not run on hardware
//
#include <hip/hip_runtime.h>

#define IN_SIZE  4096
#define OUT_SIZE 4096
#define RANK     1024

typedef __attribute__((ext_vector_type(16))) _Float16 v16h;
typedef __attribute__((ext_vector_type(8)))  _Float16 v8h;
typedef __attribute__((ext_vector_type(8)))  float    v8f;

// ---------------------------------------------------------------------------
// Dequant U:  Ut[n][r] = (int4_U[r,n] - (zero4_U[r/128, n] + 1)) * scales_U[r/128, n]
// One thread handles one packed int32 (8 consecutive r values for one n) and
// writes a single contiguous 16-byte f16 chunk (the transpose lands contiguous).
// ---------------------------------------------------------------------------
__global__ __launch_bounds__(256) void dequant_u_kernel(
    const int* __restrict__ qw, const int* __restrict__ qz,
    const float* __restrict__ sc, _Float16* __restrict__ Ut)
{
    int tid = blockIdx.x * 256 + threadIdx.x;        // 4096 * 128 threads
    int n   = tid & (OUT_SIZE - 1);                  // coalesced along n
    int r8  = tid >> 12;                             // 0..127 (packed r row)
    int g   = r8 >> 4;                               // r-group of 128
    unsigned w  = (unsigned)qw[r8 * OUT_SIZE + n];
    unsigned zq = (unsigned)qz[g * (OUT_SIZE / 8) + (n >> 3)];
    float zero = (float)(((zq >> ((n & 7) * 4)) & 15u) + 1u);
    float s    = sc[g * OUT_SIZE + n];
    v8h vals;
#pragma unroll
    for (int i = 0; i < 8; ++i) {
        float q = (float)((w >> (4 * i)) & 15u);
        vals[i] = (_Float16)((q - zero) * s);
    }
    *(v8h*)(Ut + (size_t)n * RANK + r8 * 8) = vals;  // 16B aligned store
}

// ---------------------------------------------------------------------------
// Dequant V (with S folded in):
//   Vd[k][r] = (int4_V[k,r] - (zero4_V[k/128, r] + 1)) * scales_V[k/128, r] * S[r]
// One thread handles one packed int32 (8 consecutive k, one r); writes are
// coalesced across lanes (consecutive r) for each of the 8 rows.
// ---------------------------------------------------------------------------
__global__ __launch_bounds__(256) void dequant_v_kernel(
    const int* __restrict__ qw, const int* __restrict__ qz,
    const float* __restrict__ sc, const float* __restrict__ S,
    _Float16* __restrict__ Vd)
{
    int tid = blockIdx.x * 256 + threadIdx.x;        // 512 * 1024 threads
    int r   = tid & (RANK - 1);                      // coalesced along r
    int k8  = tid >> 10;                             // 0..511 (packed k row)
    int g   = k8 >> 4;                               // k-group of 128
    unsigned w  = (unsigned)qw[k8 * RANK + r];
    unsigned zq = (unsigned)qz[g * (RANK / 8) + (r >> 3)];
    float zero = (float)(((zq >> ((r & 7) * 4)) & 15u) + 1u);
    float s    = sc[g * RANK + r] * S[r];
#pragma unroll
    for (int i = 0; i < 8; ++i) {
        float q = (float)((w >> (4 * i)) & 15u);
        Vd[(size_t)(k8 * 8 + i) * RANK + r] = (_Float16)((q - zero) * s);
    }
}

// ---------------------------------------------------------------------------
// WMMA GEMM:  out[n, k] = sum_r Ut[n, r] * Vd[k, r]
//   M = n (4096), N = k (4096), K = r (1024), f16 operands, f32 accumulate.
// Block = 256 threads = 8 waves in a 2(M) x 4(N) arrangement.
// Block tile 128x256; wave tile 64x64 = 4x4 v_wmma_f32_16x16x32_f16 tiles.
// Operands (16 MB total) are L2-resident on MI455X (192 MB L2), so fragments
// are loaded straight from global with b128 loads — no LDS staging needed.
// ---------------------------------------------------------------------------
__global__ __launch_bounds__(256) void gemm_wmma_kernel(
    const _Float16* __restrict__ A,   // Ut [4096][RANK]
    const _Float16* __restrict__ B,   // Vd [4096][RANK]
    float* __restrict__ C)            // out [4096][4096] row-major
{
    const int lane = threadIdx.x & 31;
    const int wave = threadIdx.x >> 5;
    const int Mbase = blockIdx.y * 128 + (wave >> 2) * 64;
    const int Nbase = blockIdx.x * 256 + (wave & 3) * 64;
    const int row = lane & 15;   // matrix row (A) / column (B) owned by lane
    const int hi  = lane >> 4;   // half-wave select

    // A operand per-lane K origin: 0 or 8 (elements 0-7 at +0, 8-15 at +16)
    const _Float16* ap = A + (size_t)(Mbase + row) * RANK + hi * 8;
    // B operand per-lane K origin: 0 or 8 (elements 0-7 at +0, 8-15 at +16)
    const _Float16* bp = B + (size_t)(Nbase + row) * RANK + hi * 8;

    v8f acc[4][4];
#pragma unroll
    for (int i = 0; i < 4; ++i)
#pragma unroll
        for (int j = 0; j < 4; ++j)
            acc[i][j] = (v8f){0.f, 0.f, 0.f, 0.f, 0.f, 0.f, 0.f, 0.f};

    for (int r0 = 0; r0 < RANK; r0 += 32) {
        v16h a[4], b[4];
#pragma unroll
        for (int i = 0; i < 4; ++i) {
            union { v16h v; v8h h[2]; } u;
            const _Float16* p = ap + (size_t)(16 * i) * RANK + r0;
            u.h[0] = *(const v8h*)(p);        // K kb+0 .. kb+7
            u.h[1] = *(const v8h*)(p + 16);   // K kb+16 .. kb+23
            a[i] = u.v;
        }
#pragma unroll
        for (int j = 0; j < 4; ++j) {
            union { v16h v; v8h h[2]; } u;
            const _Float16* p = bp + (size_t)(16 * j) * RANK + r0;
            u.h[0] = *(const v8h*)(p);        // K kb+0 .. kb+7
            u.h[1] = *(const v8h*)(p + 16);   // K kb+16 .. kb+23
            b[j] = u.v;
        }
#pragma unroll
        for (int i = 0; i < 4; ++i)
#pragma unroll
            for (int j = 0; j < 4; ++j)
                acc[i][j] = __builtin_amdgcn_wmma_f32_16x16x32_f16(
                    false, a[i], false, b[j], (short)0, acc[i][j],
                    false, false);
    }

    // D layout: VGPR v, col = lane&15, row = v + 8*(lane>=16)
    const int col = lane & 15;
    const int rb  = hi * 8;
#pragma unroll
    for (int i = 0; i < 4; ++i) {
#pragma unroll
        for (int j = 0; j < 4; ++j) {
            float* o = C + (size_t)(Mbase + 16 * i + rb) * IN_SIZE
                         + (Nbase + 16 * j) + col;
#pragma unroll
            for (int v = 0; v < 8; ++v)
                o[(size_t)v * IN_SIZE] = acc[i][j][v];
        }
    }
}

extern "C" void kernel_launch(void* const* d_in, const int* in_sizes, int n_in,
                              void* d_out, int out_size, void* d_ws, size_t ws_size,
                              hipStream_t stream) {
    // setup_inputs order: 0=x (dead), 1=qweight_V, 2=qzeros_V, 3=scales_V,
    // 4=g_idx_V (k/128, computed inline), 5=qweight_U, 6=qzeros_U,
    // 7=scales_U, 8=g_idx_U (inline), 9=S
    const int*   qweight_V = (const int*)d_in[1];
    const int*   qzeros_V  = (const int*)d_in[2];
    const float* scales_V  = (const float*)d_in[3];
    const int*   qweight_U = (const int*)d_in[5];
    const int*   qzeros_U  = (const int*)d_in[6];
    const float* scales_U  = (const float*)d_in[7];
    const float* S         = (const float*)d_in[9];

    _Float16* Ut = (_Float16*)d_ws;                                   // 8 MB
    _Float16* Vd = (_Float16*)((char*)d_ws +
                     (size_t)OUT_SIZE * RANK * sizeof(_Float16));     // 8 MB
    float* out = (float*)d_out;

    dequant_u_kernel<<<(OUT_SIZE * (RANK / 8)) / 256, 256, 0, stream>>>(
        qweight_U, qzeros_U, scales_U, Ut);
    dequant_v_kernel<<<((IN_SIZE / 8) * RANK) / 256, 256, 0, stream>>>(
        qweight_V, qzeros_V, scales_V, S, Vd);

    dim3 grid(OUT_SIZE / 256, IN_SIZE / 128);  // x: N(k) tiles, y: M(n) tiles
    gemm_wmma_kernel<<<grid, 256, 0, stream>>>(Ut, Vd, out);
}
